// cross_attentioni2t_single_40372692582587
// MI455X (gfx1250) — matrix operand order repaired and checked by reading, not compiled, not run on hardware
//
#include <hip/hip_runtime.h>

// ---------------------------------------------------------------------------
// Types
// ---------------------------------------------------------------------------
typedef unsigned int u32;
typedef unsigned short u16;
typedef __attribute__((ext_vector_type(4)))  u32    u32x4;
typedef __attribute__((ext_vector_type(8)))  u32    u32x8;
typedef __attribute__((ext_vector_type(4)))  int    i32x4;
typedef __attribute__((ext_vector_type(4)))  float  f32x4;
typedef __attribute__((ext_vector_type(8)))  float  v8f;
typedef __attribute__((ext_vector_type(16))) __bf16 v16bf;

#define TILE_LD 40  // bf16 elems per LDS tile row (32 data + 8 pad, conflict-free)

// --- CDNA5 async global->LDS path (guarded; falls back to sync copy) -------
#if defined(__HIP_DEVICE_COMPILE__) && defined(__gfx1250__)
#if __has_builtin(__builtin_amdgcn_global_load_async_to_lds_b128)
#define HAVE_ASYNC_LDS 1
#else
#define HAVE_ASYNC_LDS 0
#endif
#if __has_builtin(__builtin_amdgcn_s_wait_asynccnt)
#define WAIT_ASYNC() __builtin_amdgcn_s_wait_asynccnt(0)
#else
#define WAIT_ASYNC() asm volatile("s_wait_asynccnt 0" ::: "memory")
#endif
#else
#define HAVE_ASYNC_LDS 0
#define WAIT_ASYNC() ((void)0)
#endif

__device__ __forceinline__ u16 f32_to_bf16(float f) {
  u32 u = __builtin_bit_cast(u32, f);
  u32 r = u + 0x7FFFu + ((u >> 16) & 1u);  // round-to-nearest-even
  return (u16)(r >> 16);
}

// Stage 16 fp32 -> 16 bf16 into LDS (one thread's share of a 128x32 tile)
__device__ __forceinline__ void stage_f32_as_bf16(const float* __restrict__ src,
                                                  u16* dst) {
  const f32x4* s4 = (const f32x4*)src;
  u32 p[8];
#pragma unroll
  for (int i = 0; i < 4; ++i) {
    f32x4 f = s4[i];
    u16 h0 = f32_to_bf16(f[0]);
    u16 h1 = f32_to_bf16(f[1]);
    u16 h2 = f32_to_bf16(f[2]);
    u16 h3 = f32_to_bf16(f[3]);
    p[2 * i]     = (u32)h0 | ((u32)h1 << 16);
    p[2 * i + 1] = (u32)h2 | ((u32)h3 << 16);
  }
  u32x4* d = (u32x4*)dst;
  u32x4 q0 = {p[0], p[1], p[2], p[3]};
  u32x4 q1 = {p[4], p[5], p[6], p[7]};
  d[0] = q0;
  d[1] = q1;
}

// 16-byte chunk: global(bf16) -> LDS, async if available
__device__ __forceinline__ void copy16_g2l(const u16* g, u16* l) {
#if HAVE_ASYNC_LDS
  __builtin_amdgcn_global_load_async_to_lds_b128(
      (__attribute__((address_space(1))) i32x4*)g,
      (__attribute__((address_space(3))) i32x4*)l, 0, 0);
#else
  *(u32x4*)l = *(const u32x4*)g;
#endif
}

// A fragment (16x32 bf16): lane holds row m=lane&15; two 16B runs at
// k = half*8 and k = 16 + half*8   (ISA 7.12.2, 16-bit A layout)
__device__ __forceinline__ v16bf load_frag_a(const u16* lA, int m, int half) {
  u32x4 a = *(const u32x4*)(lA + m * TILE_LD + half * 8);
  u32x4 b = *(const u32x4*)(lA + m * TILE_LD + 16 + half * 8);
  u32x8 c;
  c[0] = a[0]; c[1] = a[1]; c[2] = a[2]; c[3] = a[3];
  c[4] = b[0]; c[5] = b[1]; c[6] = b[2]; c[7] = b[3];
  return __builtin_bit_cast(v16bf, c);
}

// B fragment (32x16 bf16): lane holds col n=lane&15; two 16B runs at
// k = half*8 and k = 16 + half*8  (LDS tile stored as Bt[n][k], K-contiguous)
__device__ __forceinline__ v16bf load_frag_b(const u16* lB, int n, int half) {
  u32x4 a = *(const u32x4*)(lB + n * TILE_LD + half * 8);
  u32x4 b = *(const u32x4*)(lB + n * TILE_LD + 16 + half * 8);
  u32x8 c;
  c[0] = a[0]; c[1] = a[1]; c[2] = a[2]; c[3] = a[3];
  c[4] = b[0]; c[5] = b[1]; c[6] = b[2]; c[7] = b[3];
  return __builtin_bit_cast(v16bf, c);
}

// 8 WMMAs for one 32-deep k-step of a 32x64 wave tile
__device__ __forceinline__ void wmma_step(const u16* lA, const u16* lB, int wm,
                                          int wn, int half, int l16,
                                          v8f acc[2][4]) {
  v16bf af[2], bfr[4];
#pragma unroll
  for (int mi = 0; mi < 2; ++mi)
    af[mi] = load_frag_a(lA, wm * 32 + mi * 16 + l16, half);
#pragma unroll
  for (int ni = 0; ni < 4; ++ni)
    bfr[ni] = load_frag_b(lB, wn * 64 + ni * 16 + l16, half);
#pragma unroll
  for (int mi = 0; mi < 2; ++mi)
#pragma unroll
    for (int ni = 0; ni < 4; ++ni)
      acc[mi][ni] = __builtin_amdgcn_wmma_f32_16x16x32_bf16(
          false, af[mi], false, bfr[ni], (short)0, acc[mi][ni], false, false);
}

// Stage one 128x32 bf16 A-tile and one 128x32 bf16 B-tile (bf16 sources).
// 256 threads x (2 A + 2 B) 16B chunks. gA/gB point at (tile row 0, k0).
__device__ __forceinline__ void stage_tile16(const u16* __restrict__ gA, size_t ldA,
                                             const u16* __restrict__ gB, size_t ldB,
                                             int t, u16* lA, u16* lB) {
  const int row = t >> 2;          // 0..63
  const int ch = (t & 3) * 8;      // 16B chunk within 64B row
  copy16_g2l(gA + (size_t)row * ldA + ch,        lA + row * TILE_LD + ch);
  copy16_g2l(gA + (size_t)(row + 64) * ldA + ch, lA + (row + 64) * TILE_LD + ch);
  copy16_g2l(gB + (size_t)row * ldB + ch,        lB + row * TILE_LD + ch);
  copy16_g2l(gB + (size_t)(row + 64) * ldB + ch, lB + (row + 64) * TILE_LD + ch);
}

// ---------------------------------------------------------------------------
// Projection GEMM:  Y[m, n] = bf16( sum_d X[m,d] * W[n,d] + bias[n] )
//   MODE 0: Y row-major [M, 1024]            (Q and K)
//   MODE 1: Y per-batch transposed Vt[b][n][r], m = b*256 + r   (V)
// Block 128x128, BK=32, double-buffered LDS, 8 waves (4x2), wave tile 32x64.
// ---------------------------------------------------------------------------
template <int MODE>
__global__ __launch_bounds__(256) void proj_gemm(const float* __restrict__ X,
                                                 const float* __restrict__ W,
                                                 const float* __restrict__ bias,
                                                 u16* __restrict__ Y) {
  __shared__ u16 lA0[128 * TILE_LD];
  __shared__ u16 lB0[128 * TILE_LD];
  __shared__ u16 lA1[128 * TILE_LD];
  __shared__ u16 lB1[128 * TILE_LD];
  const int t = threadIdx.x;
  const int lane = t & 31, wid = t >> 5;
  const int wm = wid & 3, wn = wid >> 2;
  const int half = lane >> 4, l16 = lane & 15;
  const int row0 = blockIdx.y * 128;
  const int col0 = blockIdx.x * 128;
  const int sr = t >> 1;            // staging row 0..127
  const int sc = (t & 1) * 16;      // staging col base {0,16}
  const float* __restrict__ Xs = X + (size_t)(row0 + sr) * 1024 + sc;
  const float* __restrict__ Ws = W + (size_t)(col0 + sr) * 1024 + sc;

  v8f acc[2][4] = {};

  stage_f32_as_bf16(Xs, &lA0[sr * TILE_LD + sc]);
  stage_f32_as_bf16(Ws, &lB0[sr * TILE_LD + sc]);
  for (int k0 = 0; k0 < 1024; k0 += 64) {
    __syncthreads();
    if (k0 + 32 < 1024) {
      stage_f32_as_bf16(Xs + k0 + 32, &lA1[sr * TILE_LD + sc]);
      stage_f32_as_bf16(Ws + k0 + 32, &lB1[sr * TILE_LD + sc]);
    }
    wmma_step(lA0, lB0, wm, wn, half, l16, acc);
    __syncthreads();
    if (k0 + 64 < 1024) {
      stage_f32_as_bf16(Xs + k0 + 64, &lA0[sr * TILE_LD + sc]);
      stage_f32_as_bf16(Ws + k0 + 64, &lB0[sr * TILE_LD + sc]);
    }
    wmma_step(lA1, lB1, wm, wn, half, l16, acc);
  }

#pragma unroll
  for (int ni = 0; ni < 4; ++ni) {
    const int n = col0 + wn * 64 + ni * 16 + l16;
    const float bv = bias[n];
#pragma unroll
    for (int mi = 0; mi < 2; ++mi) {
#pragma unroll
      for (int r = 0; r < 8; ++r) {
        const int m = row0 + wm * 32 + mi * 16 + half * 8 + r;
        const u16 h = f32_to_bf16(acc[mi][ni][r] + bv);
        if (MODE == 0) {
          Y[(size_t)m * 1024 + n] = h;
        } else {
          const int b = m >> 8, rr = m & 255;
          Y[(size_t)b * (1024 * 256) + (size_t)n * 256 + rr] = h;
        }
      }
    }
  }
}

// ---------------------------------------------------------------------------
// Generic bf16 GEMM core used by scores / out kernels (double-buffered,
// async global->LDS staging when available).  C[m,n] = sum_k A[m,k]*Bt[n,k]
// ---------------------------------------------------------------------------
__device__ __forceinline__ void bf16_gemm_core(const u16* __restrict__ gA, size_t ldA,
                                               const u16* __restrict__ gB, size_t ldB,
                                               int Kdim, int t, v8f acc[2][4],
                                               u16* lA0, u16* lB0, u16* lA1, u16* lB1) {
  const int lane = t & 31, wid = t >> 5;
  const int wm = wid & 3, wn = wid >> 2;
  const int half = lane >> 4, l16 = lane & 15;

  stage_tile16(gA, ldA, gB, ldB, t, lA0, lB0);
  for (int k0 = 0; k0 < Kdim; k0 += 64) {
    WAIT_ASYNC();
    __syncthreads();
    if (k0 + 32 < Kdim) stage_tile16(gA + k0 + 32, ldA, gB + k0 + 32, ldB, t, lA1, lB1);
    wmma_step(lA0, lB0, wm, wn, half, l16, acc);
    WAIT_ASYNC();
    __syncthreads();
    if (k0 + 64 < Kdim) stage_tile16(gA + k0 + 64, ldA, gB + k0 + 64, ldB, t, lA0, lB0);
    wmma_step(lA1, lB1, wm, wn, half, l16, acc);
  }
}

// ---------------------------------------------------------------------------
// Scores: S[b][l][r] = sum_d Q[l,d] * K[b,r,d]   (fp32 out)
// grid.x = R tile (2), grid.y = batch. Block tile 128(L) x 128(R), K=1024.
// ---------------------------------------------------------------------------
__global__ __launch_bounds__(256) void scores_gemm(const u16* __restrict__ Qb,
                                                   const u16* __restrict__ Kb,
                                                   float* __restrict__ S) {
  __shared__ u16 lA0[128 * TILE_LD];
  __shared__ u16 lB0[128 * TILE_LD];
  __shared__ u16 lA1[128 * TILE_LD];
  __shared__ u16 lB1[128 * TILE_LD];
  const int t = threadIdx.x;
  const int lane = t & 31, wid = t >> 5;
  const int wm = wid & 3, wn = wid >> 2;
  const int half = lane >> 4, l16 = lane & 15;
  const int b = blockIdx.y;
  const int col0 = blockIdx.x * 128;

  v8f acc[2][4] = {};
  bf16_gemm_core(Qb, 1024, Kb + ((size_t)b * 256 + col0) * 1024, 1024, 1024, t,
                 acc, lA0, lB0, lA1, lB1);

  float* Sb = S + (size_t)b * 128 * 256;
#pragma unroll
  for (int ni = 0; ni < 4; ++ni) {
    const int n = col0 + wn * 64 + ni * 16 + l16;
#pragma unroll
    for (int mi = 0; mi < 2; ++mi)
#pragma unroll
      for (int r = 0; r < 8; ++r) {
        const int m = wm * 32 + mi * 16 + half * 8 + r;
        Sb[(size_t)m * 256 + n] = acc[mi][ni][r];
      }
  }
}

// ---------------------------------------------------------------------------
// Softmax over R=256, then scale by 1/sqrt(D)=1/32 (reference scales AFTER
// softmax), output bf16 probabilities. One wave per row.
// ---------------------------------------------------------------------------
__global__ __launch_bounds__(256) void softmax_scale(const float* __restrict__ S,
                                                     u16* __restrict__ P) {
  const int lane = threadIdx.x & 31, wid = threadIdx.x >> 5;
  const int row = blockIdx.x * 8 + wid;  // 0 .. B*L-1
  const float* s = S + (size_t)row * 256;
  float v[8];
  float mx = -3.4e38f;
#pragma unroll
  for (int i = 0; i < 8; ++i) {
    v[i] = s[i * 32 + lane];
    mx = fmaxf(mx, v[i]);
  }
#pragma unroll
  for (int off = 16; off >= 1; off >>= 1) mx = fmaxf(mx, __shfl_xor(mx, off, 32));
  float sum = 0.f;
#pragma unroll
  for (int i = 0; i < 8; ++i) {
    v[i] = __expf(v[i] - mx);
    sum += v[i];
  }
#pragma unroll
  for (int off = 16; off >= 1; off >>= 1) sum += __shfl_xor(sum, off, 32);
  const float sc = 0.03125f / sum;  // (1/sqrt(1024)) / sum
  u16* p = P + (size_t)row * 256;
#pragma unroll
  for (int i = 0; i < 8; ++i) p[i * 32 + lane] = f32_to_bf16(v[i] * sc);
}

// ---------------------------------------------------------------------------
// Output: out[b][l][d] = sum_r P[b][l][r] * Vt[b][d][r]   (fp32 out)
// grid.x = D tile (8), grid.y = batch. Block tile 128(L) x 128(D), K=256.
// ---------------------------------------------------------------------------
__global__ __launch_bounds__(256) void out_gemm(const u16* __restrict__ P,
                                                const u16* __restrict__ Vt,
                                                float* __restrict__ O) {
  __shared__ u16 lA0[128 * TILE_LD];
  __shared__ u16 lB0[128 * TILE_LD];
  __shared__ u16 lA1[128 * TILE_LD];
  __shared__ u16 lB1[128 * TILE_LD];
  const int t = threadIdx.x;
  const int lane = t & 31, wid = t >> 5;
  const int wm = wid & 3, wn = wid >> 2;
  const int half = lane >> 4, l16 = lane & 15;
  const int b = blockIdx.y;
  const int col0 = blockIdx.x * 128;

  v8f acc[2][4] = {};
  bf16_gemm_core(P + (size_t)b * 128 * 256, 256,
                 Vt + ((size_t)b * 1024 + col0) * 256, 256, 256, t, acc, lA0,
                 lB0, lA1, lB1);

  float* Ob = O + (size_t)b * 128 * 1024;
#pragma unroll
  for (int ni = 0; ni < 4; ++ni) {
    const int n = col0 + wn * 64 + ni * 16 + l16;
#pragma unroll
    for (int mi = 0; mi < 2; ++mi)
#pragma unroll
      for (int r = 0; r < 8; ++r) {
        const int m = wm * 32 + mi * 16 + half * 8 + r;
        Ob[(size_t)m * 1024 + n] = acc[mi][ni][r];
      }
  }
}

// ---------------------------------------------------------------------------
// Launch
// ---------------------------------------------------------------------------
extern "C" void kernel_launch(void* const* d_in, const int* in_sizes, int n_in,
                              void* d_out, int out_size, void* d_ws, size_t ws_size,
                              hipStream_t stream) {
  const float* cap = (const float*)d_in[0];  // [128, 1024]
  const float* img = (const float*)d_in[1];  // [128, 256, 1024]
  const float* Wq  = (const float*)d_in[2];  // [1024, 1024]
  const float* bq  = (const float*)d_in[3];
  const float* Wk  = (const float*)d_in[4];
  const float* bk  = (const float*)d_in[5];
  const float* Wv  = (const float*)d_in[6];
  const float* bv  = (const float*)d_in[7];
  float* out = (float*)d_out;                // [128, 128, 1024]

  // Workspace carve-up (bf16 intermediates + fp32 scores)
  u16* Qb = (u16*)d_ws;                                  // 128*1024      bf16
  u16* Kb = Qb + (size_t)128 * 1024;                     // 32768*1024    bf16
  u16* Vt = Kb + (size_t)32768 * 1024;                   // [b][d][r]     bf16
  float* S = (float*)(Vt + (size_t)32768 * 1024);        // 128*128*256   f32
  u16* P  = (u16*)(S + (size_t)128 * 128 * 256);         // 128*128*256   bf16

  dim3 blk(256);
  proj_gemm<0><<<dim3(8, 1), blk, 0, stream>>>(cap, Wq, bq, Qb);
  proj_gemm<0><<<dim3(8, 256), blk, 0, stream>>>(img, Wk, bk, Kb);
  proj_gemm<1><<<dim3(8, 256), blk, 0, stream>>>(img, Wv, bv, Vt);
  scores_gemm<<<dim3(2, 128), blk, 0, stream>>>(Qb, Kb, S);
  softmax_scale<<<dim3(2048), blk, 0, stream>>>(S, P);
  out_gemm<<<dim3(8, 128), blk, 0, stream>>>(P, Vt, out);
}
